// SGB_39599598469329
// MI455X (gfx1250) — matrix operand order repaired and checked by reading, not compiled, not run on hardware
//
#include <hip/hip_runtime.h>
#include <math.h>

// ---------------- CDNA5 WMMA types ----------------
typedef __attribute__((ext_vector_type(16))) __bf16 v16bf;
typedef __attribute__((ext_vector_type(8)))  float  v8f;

union Frag { v16bf v; unsigned short s[16]; };

#define WMMA_BF16(a, b, c) \
  __builtin_amdgcn_wmma_f32_16x16x32_bf16(false, (a), false, (b), (short)0, (c), false, false)

// gfx1250 async global->LDS DMA (tracked by ASYNCcnt). lds = 32-bit LDS address
// (low bits of generic pointer per ISA LDS aperture), gaddr = 64-bit global address.
__device__ inline void async_load_b128(unsigned lds, unsigned long gaddr) {
  asm volatile("global_load_async_to_lds_b128 %0, %1, off" :: "v"(lds), "v"(gaddr)
               : "memory");
}
__device__ inline void wait_async0() {
  asm volatile("s_wait_asynccnt 0x0" ::: "memory");
}

__device__ inline unsigned short f2bf(float f) {
  union { float f; unsigned u; } v; v.f = f;
  unsigned u = v.u;
  if ((u & 0x7F800000u) == 0x7F800000u) return (unsigned short)(u >> 16); // inf/nan
  unsigned r = u + 0x7FFFu + ((u >> 16) & 1u);                            // RNE
  return (unsigned short)(r >> 16);
}
__device__ inline float sigmoidf(float x) { return 1.0f / (1.0f + expf(-x)); }

#define NEGF (-1e30f)

// ---------------- elementwise prep kernels (128-bit vectorized) ----------------
__global__ void cast_bf16x8_kernel(const float* __restrict__ src,
                                   unsigned short* __restrict__ dst, int n8) {
  int i = blockIdx.x * blockDim.x + threadIdx.x;
  if (i >= n8) return;
  const float4* s = (const float4*)src + (size_t)i * 2;
  float4 a = s[0], b = s[1];
  unsigned short o[8] = {f2bf(a.x), f2bf(a.y), f2bf(a.z), f2bf(a.w),
                         f2bf(b.x), f2bf(b.y), f2bf(b.z), f2bf(b.w)};
  *(uint4*)(dst + (size_t)i * 8) = *(const uint4*)o;
}

// Wcat[n][0..Kh) = Whh[n][:], Wcat[n][Kh..Kh+Ke) = Wih[n][:]; Kh,Ke multiples of 8
__global__ void pack_cat_kernel(const float* __restrict__ Whh,
                                const float* __restrict__ Wih,
                                unsigned short* __restrict__ dst,
                                int N, int Kh, int Ke) {
  int KC8 = (Kh + Ke) >> 3;
  int i = blockIdx.x * blockDim.x + threadIdx.x;
  if (i >= N * KC8) return;
  int n = i / KC8, k = (i % KC8) << 3;
  const float* src = (k < Kh) ? (Whh + (size_t)n * Kh + k)
                              : (Wih + (size_t)n * Ke + (k - Kh));
  const float4* s = (const float4*)src;
  float4 a = s[0], b = s[1];
  unsigned short o[8] = {f2bf(a.x), f2bf(a.y), f2bf(a.z), f2bf(a.w),
                         f2bf(b.x), f2bf(b.y), f2bf(b.z), f2bf(b.w)};
  *(uint4*)(dst + (size_t)n * (Kh + Ke) + k) = *(const uint4*)o;
}

__global__ void enc_zero_kernel(unsigned short* __restrict__ Aenc,
                                float* __restrict__ c) {
  int i = blockIdx.x * blockDim.x + threadIdx.x;
  if (i >= 64 * 64) return;
  int r = i >> 6, j = (i & 63) << 3;
  uint4 z = {0u, 0u, 0u, 0u};
  *(uint4*)(Aenc + (size_t)r * 768 + j) = z;                 // h part
  float4 zf = {0.f, 0.f, 0.f, 0.f};
  ((float4*)(c + (size_t)r * 512 + j))[0] = zf;
  ((float4*)(c + (size_t)r * 512 + j))[1] = zf;
}

// encoder input token at step t: t==0 -> PART(0); fwd: s[b][t-1]; bwd: s[b][64-t]
__global__ void enc_gather_kernel(const int* __restrict__ sent,
                                  const unsigned short* __restrict__ emb,
                                  unsigned short* __restrict__ Aenc, int t) {
  int i = blockIdx.x * blockDim.x + threadIdx.x;   // 64 * 32
  if (i >= 64 * 32) return;
  int r = i >> 5, e = (i & 31) << 3;
  int b = r & 31;
  int tok = 0;
  if (t != 0) tok = (r < 32) ? sent[b * 64 + (t - 1)] : sent[b * 64 + (64 - t)];
  *(uint4*)(Aenc + (size_t)r * 768 + 512 + e) =
      *(const uint4*)(emb + (size_t)tok * 256 + e);
}

__global__ void dec_init_kernel(const unsigned short* __restrict__ hsBf,
                                const float* __restrict__ hsF,
                                unsigned short* __restrict__ Adec,
                                float* __restrict__ cdec, int dir) {
  int i = blockIdx.x * blockDim.x + threadIdx.x;   // 2048 * 64
  if (i >= 2048 * 64) return;
  int rb = i >> 6, j = (i & 63) << 3;
  int t = rb >> 5, b = rb & 31;
  int src = t * 64 + (dir ? 32 + b : b);
  *(uint4*)(Adec + (size_t)rb * 768 + j) = *(const uint4*)(hsBf + (size_t)src * 512 + j);
  const float4* cs = (const float4*)(hsF + (size_t)src * 512 + j);
  float4* cd = (float4*)(cdec + (size_t)rb * 512 + j);
  cd[0] = cs[0];
  cd[1] = cs[1];
}

// decoder input token at step s: s==0 -> PART; else seg char j=s-1 at pos=min(t+s-1,63)
__global__ void dec_gather_kernel(const int* __restrict__ sent,
                                  const unsigned short* __restrict__ emb,
                                  unsigned short* __restrict__ Adec, int s, int dir) {
  int i = blockIdx.x * blockDim.x + threadIdx.x;   // 2048 * 32
  if (i >= 2048 * 32) return;
  int rb = i >> 5, e = (i & 31) << 3;
  int t = rb >> 5, b = rb & 31;
  int tok = 0;
  if (s != 0) {
    int pos = t + s - 1; if (pos > 63) pos = 63;
    tok = dir ? sent[b * 64 + (63 - pos)] : sent[b * 64 + pos];
  }
  *(uint4*)(Adec + (size_t)rb * 768 + 512 + e) =
      *(const uint4*)(emb + (size_t)tok * 256 + e);
}

// target char for logits row m = s*2048 + rb  (s<3: char at pos=min(t+s,63); else -1)
__global__ void tgt_fill_kernel(const int* __restrict__ sent,
                                int* __restrict__ tgt, int dir) {
  int i = blockIdx.x * blockDim.x + threadIdx.x;
  if (i >= 8192) return;
  int s = i >> 11, rb = i & 2047;
  int t = rb >> 5, b = rb & 31;
  int tok = -1;
  if (s < 3) {
    int pos = t + s; if (pos > 63) pos = 63;
    tok = dir ? sent[b * 64 + (63 - pos)] : sent[b * 64 + pos];
  }
  tgt[i] = tok;
}

// ---------------- generic bf16 WMMA GEMM: C = A[MxK] * W[NxK]^T (+bias) --------------
// block: 256 threads = 8 waves, tile 64(M) x 128(N); each wave owns a 32x32 register
// tile (2x2 accumulators -> each LDS fragment reused twice). K chunk 64.
// Tile staging uses gfx1250 async global->LDS DMA (ASYNCcnt), no VGPR round-trip.
// REQUIRES: M % 64 == 0, N % 128 == 0, K % 64 == 0 (true for all call sites).
__global__ void gemm_bf16_wmma(const unsigned short* __restrict__ A,
                               const unsigned short* __restrict__ Wt,
                               const float* __restrict__ bias,
                               float* __restrict__ C,
                               int M, int N, int K) {
  __shared__ __align__(16) unsigned short As[64][64];    // 8 KB
  __shared__ __align__(16) unsigned short Ws[128][64];   // 16 KB
  int tid = threadIdx.x;
  int wave = tid >> 5, lane = tid & 31;
  int m0 = blockIdx.y * 64;
  int n0 = blockIdx.x * 128;
  int wm = (wave >> 2) * 32;   // 0 or 32
  int wn = (wave & 3) * 32;    // 0..96
  int row = lane & 15, hi = lane >> 4;
  v8f z = {0.f, 0.f, 0.f, 0.f, 0.f, 0.f, 0.f, 0.f};
  v8f acc00 = z, acc01 = z, acc10 = z, acc11 = z;

  int ar = tid >> 3, ac = (tid & 7) << 3;   // A: 64 rows x 8 chunks = 512 vec loads
  unsigned ldsA0 = (unsigned)(size_t)&As[ar][ac];
  unsigned ldsA1 = (unsigned)(size_t)&As[ar + 32][ac];
  unsigned ldsW0 = (unsigned)(size_t)&Ws[ar][ac];

  for (int k0 = 0; k0 < K; k0 += 64) {
    async_load_b128(ldsA0, (unsigned long)(A + (size_t)(m0 + ar) * K + k0 + ac));
    async_load_b128(ldsA1, (unsigned long)(A + (size_t)(m0 + ar + 32) * K + k0 + ac));
#pragma unroll
    for (int j = 0; j < 4; ++j) {
      int r = ar + 32 * j;
      async_load_b128(ldsW0 + (unsigned)(32 * j * 64 * 2),
                      (unsigned long)(Wt + (size_t)(n0 + r) * K + k0 + ac));
    }
    if (k0 + 64 < K) {          // gfx1250 global_prefetch for next K phase
      __builtin_prefetch(A + (size_t)(m0 + ar) * K + k0 + 64 + ac, 0, 1);
      __builtin_prefetch(Wt + (size_t)(n0 + ar) * K + k0 + 64 + ac, 0, 1);
      __builtin_prefetch(Wt + (size_t)(n0 + ar + 64) * K + k0 + 64 + ac, 0, 1);
    }
    wait_async0();
    __syncthreads();
#pragma unroll
    for (int kk = 0; kk < 64; kk += 32) {
      Frag a0, a1, b0, b1;
#pragma unroll
      for (int i = 0; i < 16; ++i) {        // A: lane = row, K striped by lane-half
        int g = i >> 3, w2 = i & 7;
        int kc = kk + hi * 16 + i;
        a0.s[i] = As[wm + row][kc];
        a1.s[i] = As[wm + 16 + row][kc];
      }
#pragma unroll
      for (int i = 0; i < 16; ++i) {        // B: lane = column, K striped by lane-half
        b0.s[i] = Ws[wn + row][kk + hi * 16 + i];
        b1.s[i] = Ws[wn + 16 + row][kk + hi * 16 + i];
      }
      acc00 = WMMA_BF16(a0.v, b0.v, acc00);
      acc01 = WMMA_BF16(a0.v, b1.v, acc01);
      acc10 = WMMA_BF16(a1.v, b0.v, acc10);
      acc11 = WMMA_BF16(a1.v, b1.v, acc11);
    }
    __syncthreads();
  }
#pragma unroll
  for (int r = 0; r < 8; ++r) {             // C: VGPR r -> M=r(+8 hi), N=lane%16
    int gm0 = m0 + wm + r + 8 * hi;
    int gm1 = gm0 + 16;
    int gn0 = n0 + wn + row;
    int gn1 = gn0 + 16;
    float bi0 = bias[gn0], bi1 = bias[gn1];
    C[(size_t)gm0 * N + gn0] = acc00[r] + bi0;
    C[(size_t)gm0 * N + gn1] = acc01[r] + bi1;
    C[(size_t)gm1 * N + gn0] = acc10[r] + bi0;
    C[(size_t)gm1 * N + gn1] = acc11[r] + bi1;
  }
}

// ---------------- LSTM pointwise (gate order i,f,g,o) ----------------
__global__ void lstm_pointwise(const float* __restrict__ gates,  // [M,4H]
                               float* __restrict__ c,            // [M,H]
                               unsigned short* __restrict__ hA,  // act buf h-part
                               int strideA,
                               unsigned short* __restrict__ hsBf,// optional [M,H]
                               float* __restrict__ hsF,          // optional [M,H]
                               int M, int H) {
  int i = blockIdx.x * blockDim.x + threadIdx.x;
  if (i >= M * H) return;
  int r = i / H, j = i % H;
  const float* g = gates + (size_t)r * 4 * H;
  float gi = sigmoidf(g[j]);
  float gf = sigmoidf(g[H + j]);
  float gg = tanhf(g[2 * H + j]);
  float go = sigmoidf(g[3 * H + j]);
  float cv = gf * c[i] + gi * gg;
  c[i] = cv;
  float h = go * tanhf(cv);
  unsigned short hb = f2bf(h);
  hA[(size_t)r * strideA + j] = hb;
  if (hsBf) hsBf[i] = hb;
  if (hsF)  hsF[i]  = h;
}

// ---------------- fused logits GEMM + online logsumexp ----------------
// A: [Mrows,512] bf16 hidden states, Wt: [V,512] bf16, hb: [V] f32.
// Per row: logZ, logit[tgt], logit[0] (PART). Block = 16 rows; each wave covers two
// 16-col tiles (A fragments reused) -> 256 columns per chunk, 32 chunks over V=8000.
// f32 staging tile aliases Ws (Ws is dead between last K phase and next chunk reload).
__global__ void lse_gemm_wmma(const unsigned short* __restrict__ A,
                              const unsigned short* __restrict__ Wt,
                              const float* __restrict__ hb,
                              const int* __restrict__ tgt,
                              float* __restrict__ logZ,
                              float* __restrict__ tgtLogit,
                              float* __restrict__ partLogit,
                              int V) {
  __shared__ __align__(16) unsigned short As[16][512];   // 16 KB
  __shared__ __align__(16) unsigned short Ws[256][64];   // 32 KB (tile aliases this)
  __shared__ float runm[16], runs[16];
  __shared__ float redm[16][16], reds[16][16];
  __shared__ float capT[16], capP[16];
  __shared__ int tgts[16];
  float (*tile)[16][16] = (float (*)[16][16])&Ws[0][0];  // [16][16][16] = 16 KB

  int tid = threadIdx.x, wave = tid >> 5, lane = tid & 31;
  int m0 = blockIdx.x * 16;
  {                                        // A: 16x512, 4 vec loads per thread
    uint4 ta[4];
#pragma unroll
    for (int j = 0; j < 4; ++j) {
      int v = tid + 256 * j;
      int r = v >> 6, c = (v & 63) << 3;
      ta[j] = *(const uint4*)(A + (size_t)(m0 + r) * 512 + c);
    }
#pragma unroll
    for (int j = 0; j < 4; ++j) {
      int v = tid + 256 * j;
      int r = v >> 6, c = (v & 63) << 3;
      *(uint4*)(&As[r][c]) = ta[j];
    }
  }
  if (tid < 16) {
    runm[tid] = -3.0e38f; runs[tid] = 0.0f;
    tgts[tid] = tgt[m0 + tid];
    capT[tid] = 0.0f; capP[tid] = 0.0f;
  }
  __syncthreads();

  int row = lane & 15, hi = lane >> 4;
  int wr = tid >> 3, wc = (tid & 7) << 3;  // Ws: 256 rows x 8 chunks, 8 per thread
  int nchunks = (V + 255) / 256;
  for (int nc = 0; nc < nchunks; ++nc) {
    int n0 = nc * 256;
    v8f z = {0.f, 0.f, 0.f, 0.f, 0.f, 0.f, 0.f, 0.f};
    v8f acc0 = z, acc1 = z;
    for (int kc = 0; kc < 512; kc += 64) {
      uint4 tv[8];                         // load all 8 first (MLP), then store
#pragma unroll
      for (int j = 0; j < 8; ++j) {
        int gn = n0 + wr + 32 * j;
        uint4 val = {0u, 0u, 0u, 0u};
        if (gn < V) val = *(const uint4*)(Wt + (size_t)gn * 512 + kc + wc);
        tv[j] = val;
      }
#pragma unroll
      for (int j = 0; j < 8; ++j)
        *(uint4*)(&Ws[wr + 32 * j][wc]) = tv[j];
      __syncthreads();
#pragma unroll
      for (int kk = 0; kk < 64; kk += 32) {
        Frag a, b0, b1;
#pragma unroll
        for (int i = 0; i < 16; ++i) {
          int g = i >> 3, w2 = i & 7;
          a.s[i] = As[row][kc + kk + hi * 16 + i];
        }
#pragma unroll
        for (int i = 0; i < 16; ++i) {
          b0.s[i] = Ws[wave * 32 + row][kk + hi * 16 + i];
          b1.s[i] = Ws[wave * 32 + 16 + row][kk + hi * 16 + i];
        }
        acc0 = WMMA_BF16(a.v, b0.v, acc0);
        acc1 = WMMA_BF16(a.v, b1.v, acc1);
      }
      __syncthreads();
    }
    // stage logits (+hb) for this 16x256 chunk (overwrites Ws region - now dead)
#pragma unroll
    for (int r = 0; r < 8; ++r) {
      int gr = r + 8 * hi;
      int g0 = n0 + wave * 32 + row;
      int g1 = g0 + 16;
      tile[wave * 2][gr][row]     = (g0 < V) ? (acc0[r] + hb[g0]) : -3.0e38f;
      tile[wave * 2 + 1][gr][row] = (g1 < V) ? (acc1[r] + hb[g1]) : -3.0e38f;
    }
    __syncthreads();
    // per-row online logsumexp merge (16 rows x 16 col-threads)
    int rr = tid >> 4, cc = tid & 15;
    float lm = -3.0e38f;
    for (int wt = 0; wt < 16; ++wt) {
      float v = tile[wt][rr][cc];
      lm = fmaxf(lm, v);
      int gcol = n0 + wt * 16 + cc;
      if (gcol < V) {
        if (gcol == 0) capP[rr] = v;
        if (gcol == tgts[rr]) capT[rr] = v;
      }
    }
    redm[rr][cc] = lm;
    __syncthreads();
    for (int st = 8; st > 0; st >>= 1) {
      if (cc < st) redm[rr][cc] = fmaxf(redm[rr][cc], redm[rr][cc + st]);
      __syncthreads();
    }
    float cm = redm[rr][0];
    float ls = 0.0f;
    for (int wt = 0; wt < 16; ++wt) ls += expf(tile[wt][rr][cc] - cm);
    reds[rr][cc] = ls;
    __syncthreads();
    for (int st = 8; st > 0; st >>= 1) {
      if (cc < st) reds[rr][cc] += reds[rr][cc + st];
      __syncthreads();
    }
    if (cc == 0) {
      float cs = reds[rr][0];
      float m = runm[rr], s = runs[rr];
      float nm = fmaxf(m, cm);
      runs[rr] = s * expf(m - nm) + cs * expf(cm - nm);
      runm[rr] = nm;
    }
    __syncthreads();
  }
  if (tid < 16) {
    logZ[m0 + tid] = runm[tid] + logf(runs[tid]);
    tgtLogit[m0 + tid] = capT[tid];
    partLogit[m0 + tid] = capP[tid];
  }
}

// ---------------- final: segment scores + semi-Markov DP + loss ----------------
__device__ inline float seg_res(const float* __restrict__ tl,
                                const float* __restrict__ lz,
                                const float* __restrict__ pl,
                                int t, int j, int b) {
  int rb = t * 32 + b;
  float cum = 0.0f;
  for (int i = 0; i <= j; ++i)
    cum += tl[i * 2048 + rb] - lz[i * 2048 + rb];
  cum += pl[(j + 1) * 2048 + rb] - lz[(j + 1) * 2048 + rb];
  return cum;
}

__global__ void final_dp_kernel(const float* lzF, const float* tlF, const float* plF,
                                const float* lzB, const float* tlB, const float* plB,
                                float* __restrict__ out) {
  __shared__ float red[32];
  int b = threadIdx.x;
  float buf0 = 0.0f, buf1 = NEGF, buf2 = NEGF;  // buf[d-1] = total[j-d]
  for (int e = 0; e < 64; ++e) {
    float terms[3];
    float bufs[3] = {buf0, buf1, buf2};
    for (int l = 0; l < 3; ++l) {
      int st = e - l;
      float p;
      if (st >= 0)
        p = seg_res(tlF, lzF, plF, st, l, b) + seg_res(tlB, lzB, plB, 63 - e, l, b);
      else
        p = NEGF;
      terms[l] = bufs[l] + p;
    }
    float mx = fmaxf(terms[0], fmaxf(terms[1], terms[2]));
    float tot = mx + logf(expf(terms[0] - mx) + expf(terms[1] - mx) + expf(terms[2] - mx));
    buf2 = buf1; buf1 = buf0; buf0 = tot;
  }
  red[b] = buf0;
  __syncthreads();
  if (b == 0) {
    float sm = 0.0f;
    for (int i = 0; i < 32; ++i) sm += red[i];
    out[0] = -sm / 32.0f;
  }
}

// ---------------- host orchestration ----------------
extern "C" void kernel_launch(void* const* d_in, const int* in_sizes, int n_in,
                              void* d_out, int out_size, void* d_ws, size_t ws_size,
                              hipStream_t stream) {
  (void)in_sizes; (void)n_in; (void)out_size; (void)ws_size;
  const int*   sent = (const int*)d_in[0];
  const float* emb  = (const float*)d_in[1];
  const float* eWih = (const float*)d_in[2];
  const float* eWhh = (const float*)d_in[3];
  const float* eb   = (const float*)d_in[4];
  const float* fWih = (const float*)d_in[5];
  const float* fWhh = (const float*)d_in[6];
  const float* fb   = (const float*)d_in[7];
  const float* fhW  = (const float*)d_in[8];
  const float* fhb  = (const float*)d_in[9];
  const float* bWih = (const float*)d_in[10];
  const float* bWhh = (const float*)d_in[11];
  const float* bb   = (const float*)d_in[12];
  const float* bhW  = (const float*)d_in[13];
  const float* bhb  = (const float*)d_in[14];

  const int T = 64, V = 8000, E = 256, H = 512, G = 2048, KC = 768;
  const int DB = 2048, MR = 8192;

  char* ws = (char*)d_ws;
  size_t off = 0;
  auto alloc = [&](size_t bytes) -> void* {
    off = (off + 255) & ~(size_t)255;
    void* p = ws + off;
    off += bytes;
    return p;
  };

  unsigned short* embBf = (unsigned short*)alloc((size_t)V * E * 2);
  unsigned short* catE  = (unsigned short*)alloc((size_t)G * KC * 2);
  unsigned short* catF  = (unsigned short*)alloc((size_t)G * KC * 2);
  unsigned short* catB  = (unsigned short*)alloc((size_t)G * KC * 2);
  unsigned short* hWfBf = (unsigned short*)alloc((size_t)V * H * 2);
  unsigned short* hWbBf = (unsigned short*)alloc((size_t)V * H * 2);
  unsigned short* Aenc  = (unsigned short*)alloc((size_t)64 * KC * 2);
  float*          cEnc  = (float*)alloc((size_t)64 * H * 4);
  float*          gEnc  = (float*)alloc((size_t)64 * G * 4);
  unsigned short* hsBf  = (unsigned short*)alloc((size_t)T * 64 * H * 2);
  float*          hsF   = (float*)alloc((size_t)T * 64 * H * 4);
  unsigned short* Adec  = (unsigned short*)alloc((size_t)DB * KC * 2);
  float*          cDec  = (float*)alloc((size_t)DB * H * 4);
  float*          gDec  = (float*)alloc((size_t)DB * G * 4);
  unsigned short* HS    = (unsigned short*)alloc((size_t)4 * DB * H * 2);
  int* tgtF = (int*)alloc((size_t)MR * 4);
  int* tgtB = (int*)alloc((size_t)MR * 4);
  float* lzF = (float*)alloc((size_t)MR * 4);
  float* tlF = (float*)alloc((size_t)MR * 4);
  float* plF = (float*)alloc((size_t)MR * 4);
  float* lzB = (float*)alloc((size_t)MR * 4);
  float* tlB = (float*)alloc((size_t)MR * 4);
  float* plB = (float*)alloc((size_t)MR * 4);

  auto gemm = [&](const unsigned short* A, const unsigned short* Wt,
                  const float* bias, float* C, int M, int N, int K) {
    dim3 g(N / 128, M / 64);
    gemm_bf16_wmma<<<g, 256, 0, stream>>>(A, Wt, bias, C, M, N, K);
  };

  // weight prep (all sizes are multiples of 8)
  cast_bf16x8_kernel<<<(V * E / 8 + 255) / 256, 256, 0, stream>>>(emb, embBf, V * E / 8);
  pack_cat_kernel<<<(G * KC / 8 + 255) / 256, 256, 0, stream>>>(eWhh, eWih, catE, G, H, E);
  pack_cat_kernel<<<(G * KC / 8 + 255) / 256, 256, 0, stream>>>(fWhh, fWih, catF, G, H, E);
  pack_cat_kernel<<<(G * KC / 8 + 255) / 256, 256, 0, stream>>>(bWhh, bWih, catB, G, H, E);
  cast_bf16x8_kernel<<<(V * H / 8 + 255) / 256, 256, 0, stream>>>(fhW, hWfBf, V * H / 8);
  cast_bf16x8_kernel<<<(V * H / 8 + 255) / 256, 256, 0, stream>>>(bhW, hWbBf, V * H / 8);

  // encoder (fwd rows 0..31, bwd rows 32..63 batched; shared weights)
  enc_zero_kernel<<<(64 * 64 + 255) / 256, 256, 0, stream>>>(Aenc, cEnc);
  for (int t = 0; t < T; ++t) {
    enc_gather_kernel<<<(64 * 32 + 255) / 256, 256, 0, stream>>>(sent, embBf, Aenc, t);
    gemm(Aenc, catE, eb, gEnc, 64, G, KC);
    lstm_pointwise<<<(64 * H + 255) / 256, 256, 0, stream>>>(
        gEnc, cEnc, Aenc, KC,
        hsBf + (size_t)t * 64 * H, hsF + (size_t)t * 64 * H, 64, H);
  }

  // decoders (dir 0 = fwd, dir 1 = bwd)
  for (int dir = 0; dir < 2; ++dir) {
    const unsigned short* cat = dir ? catB : catF;
    const float* gb  = dir ? bb : fb;
    const unsigned short* hw = dir ? hWbBf : hWfBf;
    const float* hbias = dir ? bhb : fhb;
    int*   tg = dir ? tgtB : tgtF;
    float* lz = dir ? lzB : lzF;
    float* tl = dir ? tlB : tlF;
    float* pl = dir ? plB : plF;

    dec_init_kernel<<<(DB * 64 + 255) / 256, 256, 0, stream>>>(hsBf, hsF, Adec, cDec, dir);
    for (int s = 0; s < 4; ++s) {
      dec_gather_kernel<<<(DB * 32 + 255) / 256, 256, 0, stream>>>(sent, embBf, Adec, s, dir);
      gemm(Adec, cat, gb, gDec, DB, G, KC);
      lstm_pointwise<<<(DB * H + 255) / 256, 256, 0, stream>>>(
          gDec, cDec, Adec, KC, HS + (size_t)s * DB * H, nullptr, DB, H);
    }
    tgt_fill_kernel<<<(MR + 255) / 256, 256, 0, stream>>>(sent, tg, dir);
    lse_gemm_wmma<<<MR / 16, 256, 0, stream>>>(HS, hw, hbias, tg, lz, tl, pl, V);
  }

  final_dp_kernel<<<1, 32, 0, stream>>>(lzF, tlF, plF, lzB, tlB, plB, (float*)d_out);
}
